// MHAttnWithBias_34531537060006
// MI455X (gfx1250) — matrix operand order repaired and checked by reading, not compiled, not run on hardware
//
#include <hip/hip_runtime.h>
#include <hip/hip_bf16.h>
#include <math.h>
#include <stdint.h>

typedef __attribute__((ext_vector_type(16))) _Float16 v16h;
typedef __attribute__((ext_vector_type(8)))  float    v8f;

#define D_MODEL 2048
#define NHEAD   16
#define DK      128
#define NQ      2048
#define NK      2048

__device__ __forceinline__ v8f wmma_f16(const v16h& a, const v16h& b, const v8f& c) {
  // 8 args: (neg_a, A, neg_b, B, c_mod, C, reuse_a, reuse_b)
  return __builtin_amdgcn_wmma_f32_16x16x32_f16(false, a, false, b, (short)0, c, false, false);
}

// gfx1250 async copy: 16B global -> LDS, tracked by ASYNCcnt (no VGPR data path).
__device__ __forceinline__ void async_cp16(const void* gptr, void* ldsptr) {
  const uint32_t lds_off = (uint32_t)(uintptr_t)ldsptr;           // low 32b of generic = LDS offset
  const unsigned long long ga = (unsigned long long)(uintptr_t)gptr;
  asm volatile("global_load_async_to_lds_b128 %0, %1, off"
               :: "v"(lds_off), "v"(ga) : "memory");
}
__device__ __forceinline__ void async_wait0() {
  asm volatile("s_wait_asynccnt 0x0" ::: "memory");
}

// ---------------------------------------------------------------------------
// Y[n][m] = sum_k X[n][k] * W[m][k] + b[m]
// 64x64 tile per 128-thread block (4 waves), k-step 64, float4 global loads,
// f16 LDS tiles, v_wmma_f32_16x16x32_f16 with fp32 accumulation.
// ---------------------------------------------------------------------------
template <bool OUT_F16>
__global__ __launch_bounds__(128)
void proj_gemm(const float* __restrict__ X, const float* __restrict__ W,
               const float* __restrict__ bvec, void* __restrict__ Yout) {
  __shared__ __align__(16) _Float16 As[64][64];
  __shared__ __align__(16) _Float16 Bs[64][64];
  const int tid  = threadIdx.x;
  const int lane = tid & 31, wave = tid >> 5;
  const int g = lane >> 4, ln = lane & 15;
  const int m0 = blockIdx.x * 64;  // output columns
  const int n0 = blockIdx.y * 64;  // output rows

  v8f acc[4] = {};

  for (int k0 = 0; k0 < D_MODEL; k0 += 64) {
    // 64x64 f32 per array = 1024 float4 chunks; 8 per thread per array
    for (int t = tid; t < 1024; t += 128) {
      const int r = t >> 4, c4 = (t & 15) << 2;
      const float4 x = *(const float4*)&X[(size_t)(n0 + r) * D_MODEL + k0 + c4];
      As[r][c4 + 0] = (_Float16)x.x;  As[r][c4 + 1] = (_Float16)x.y;
      As[r][c4 + 2] = (_Float16)x.z;  As[r][c4 + 3] = (_Float16)x.w;
      const float4 w = *(const float4*)&W[(size_t)(m0 + r) * D_MODEL + k0 + c4];
      Bs[r][c4 + 0] = (_Float16)w.x;  Bs[r][c4 + 1] = (_Float16)w.y;
      Bs[r][c4 + 2] = (_Float16)w.z;  Bs[r][c4 + 3] = (_Float16)w.w;
    }
    __syncthreads();

#pragma unroll
    for (int c = 0; c < 2; ++c) {
      v16h a;
#pragma unroll
      for (int i = 0; i < 16; ++i) {
        const int kl = i + 16 * g;
        a[i] = As[wave * 16 + ln][c * 32 + kl];
      }
#pragma unroll
      for (int j = 0; j < 4; ++j) {
        v16h b;
#pragma unroll
        for (int i = 0; i < 16; ++i)
          b[i] = Bs[j * 16 + ln][c * 32 + i + 16 * g];  // B[k][n] = W[m0+j*16+n][k]
        acc[j] = wmma_f16(a, b, acc[j]);
      }
    }
    __syncthreads();
  }

#pragma unroll
  for (int j = 0; j < 4; ++j) {
    const int col = m0 + j * 16 + ln;
    const float bb = bvec[col];
#pragma unroll
    for (int r = 0; r < 8; ++r) {
      const int row = n0 + wave * 16 + r + 8 * g;
      const float v = acc[j][r] + bb;
      if (OUT_F16)
        ((_Float16*)Yout)[(size_t)row * D_MODEL + col] = (_Float16)v;
      else
        ((float*)Yout)[(size_t)row * D_MODEL + col] = v;
    }
  }
}

// ---------------------------------------------------------------------------
// Flash attention: block = (64 q-rows, 1 head), 4 waves x 16 q-rows each.
// Double-buffered K/V LDS tiles filled with global_load_async_to_lds_b128,
// online softmax over 32-key chunks, S and O GEMMs via WMMA f16,
// bias stream prefetched one chunk ahead.
// ---------------------------------------------------------------------------
__global__ __launch_bounds__(128)
void attn_kernel(const _Float16* __restrict__ Qh, const _Float16* __restrict__ Kh,
                 const _Float16* __restrict__ Vh, const float* __restrict__ bias,
                 float* __restrict__ O) {
  __shared__ __align__(16) _Float16 Ks[2][32][DK];
  __shared__ __align__(16) _Float16 Vs[2][32][DK];
  __shared__ __align__(16) _Float16 Ps[4][16][32];

  const int tid  = threadIdx.x;
  const int lane = tid & 31, wave = tid >> 5;
  const int g = lane >> 4, ln = lane & 15;
  const int h  = blockIdx.y;
  const int q0 = blockIdx.x * 64 + wave * 16;
  const float scale = 0.08838834764831845f;  // 1/sqrt(128)

  // stage one 32x128 f16 chunk of Kh and Vh into buffer `buf` (async, 16B units)
  auto stage = [&](int buf, int kc) {
    // per array: 32 rows * 16 chunks = 512; two arrays = 1024; 8 per thread
    for (int t = tid; t < 1024; t += 128) {
      const int arr = t >> 9;
      const int e   = t & 511;
      const int r   = e >> 4;
      const int c   = (e & 15) << 3;  // half-index, 8 halves = 16B
      const _Float16* gsrc = (arr ? Vh : Kh) + (size_t)(kc + r) * D_MODEL + h * DK + c;
      _Float16* ldst = arr ? &Vs[buf][r][c] : &Ks[buf][r][c];
      async_cp16(gsrc, ldst);
    }
  };

  // Q A-fragments for this wave's 16 rows, 4 dk-chunks of 32
  v16h qf[4];
#pragma unroll
  for (int c = 0; c < 4; ++c)
#pragma unroll
    for (int i = 0; i < 16; ++i) {
      const int kl = i + 16 * g;
      qf[c][i] = Qh[(size_t)(q0 + ln) * D_MODEL + h * DK + c * 32 + kl];
    }

  v8f o[8] = {};
  float mrow[8], lrow[8];
#pragma unroll
  for (int r = 0; r < 8; ++r) { mrow[r] = -1e30f; lrow[r] = 0.0f; }

  stage(0, 0);
  async_wait0();
  __syncthreads();

  for (int kc = 0, it = 0; kc < NK; kc += 32, ++it) {
    const int buf = it & 1;
    if (kc + 32 < NK) {
      stage(buf ^ 1, kc + 32);  // overlap next chunk's staging with compute
#pragma unroll
      for (int r = 0; r < 8; ++r)  // run the bias HBM stream one chunk ahead
        __builtin_prefetch(&bias[((size_t)h * NQ + q0 + r + 8 * g) * NK + kc + 32 + ln], 0, 1);
    }

    // S(16q x 32key) = Q(16x128) . K_chunk^T
    v8f s0 = {}, s1 = {};
#pragma unroll
    for (int c = 0; c < 4; ++c) {
      v16h b0, b1;
#pragma unroll
      for (int i = 0; i < 16; ++i) {
        b0[i] = Ks[buf][ln][c * 32 + i + 16 * g];       // B[k=dk][n=key 0..15]
        b1[i] = Ks[buf][16 + ln][c * 32 + i + 16 * g];  // keys 16..31
      }
      s0 = wmma_f16(qf[c], b0, s0);
      s1 = wmma_f16(qf[c], b1, s1);
    }

    // bias + scale + online softmax update (row = r + 8g, col = ln)
#pragma unroll
    for (int r = 0; r < 8; ++r) {
      const int qrow = q0 + r + 8 * g;
      float v0 = s0[r] * scale + bias[((size_t)h * NQ + qrow) * NK + kc + ln];
      float v1 = s1[r] * scale + bias[((size_t)h * NQ + qrow) * NK + kc + 16 + ln];
      float tmax = fmaxf(v0, v1);
#pragma unroll
      for (int off = 1; off < 16; off <<= 1)
        tmax = fmaxf(tmax, __shfl_xor(tmax, off, 32));
      const float mnew  = fmaxf(mrow[r], tmax);
      const float alpha = __expf(mrow[r] - mnew);
      const float p0 = __expf(v0 - mnew);
      const float p1 = __expf(v1 - mnew);
      float psum = p0 + p1;
#pragma unroll
      for (int off = 1; off < 16; off <<= 1)
        psum += __shfl_xor(psum, off, 32);
      lrow[r] = lrow[r] * alpha + psum;
      mrow[r] = mnew;
#pragma unroll
      for (int j = 0; j < 8; ++j) o[j][r] *= alpha;
      Ps[wave][r + 8 * g][ln]      = (_Float16)p0;
      Ps[wave][r + 8 * g][16 + ln] = (_Float16)p1;
    }
    __syncthreads();  // make cross-lane P visible before A-fragment gather

    // P as A-fragment (16 x 32 keys)
    v16h pa;
#pragma unroll
    for (int i = 0; i < 16; ++i) {
      const int kl = i + 16 * g;
      pa[i] = Ps[wave][ln][kl];
    }
    // O(16 x 128) += P(16x32) . V_chunk(32x128)
#pragma unroll
    for (int j = 0; j < 8; ++j) {
      v16h vb;
#pragma unroll
      for (int i = 0; i < 16; ++i)
        vb[i] = Vs[buf][i + 16 * g][j * 16 + ln];  // B[k=key][n=dk col]
      o[j] = wmma_f16(pa, vb, o[j]);
    }

    async_wait0();    // next chunk's async fills must land before anyone reads them
    __syncthreads();  // and before anyone re-stages over the buffer we just read
  }

  // normalize and write fp32 merged-head layout [q][h*DK + d]
#pragma unroll
  for (int j = 0; j < 8; ++j)
#pragma unroll
    for (int r = 0; r < 8; ++r) {
      const int qrow = q0 + r + 8 * g;
      O[(size_t)qrow * D_MODEL + h * DK + j * 16 + ln] = o[j][r] / lrow[r];
    }
}

// ---------------------------------------------------------------------------
extern "C" void kernel_launch(void* const* d_in, const int* in_sizes, int n_in,
                              void* d_out, int out_size, void* d_ws, size_t ws_size,
                              hipStream_t stream) {
  const float* Q    = (const float*)d_in[0];
  const float* K    = (const float*)d_in[1];
  const float* V    = (const float*)d_in[2];
  const float* bias = (const float*)d_in[3];
  const float* Wq   = (const float*)d_in[4];
  const float* bq   = (const float*)d_in[5];
  const float* Wk   = (const float*)d_in[6];
  const float* bk   = (const float*)d_in[7];
  const float* Wv   = (const float*)d_in[8];
  const float* bv   = (const float*)d_in[9];
  const float* Wo   = (const float*)d_in[10];
  const float* bo   = (const float*)d_in[11];

  char* ws = (char*)d_ws;
  _Float16* Qh = (_Float16*)(ws);                                 // 8 MB
  _Float16* Kh = (_Float16*)(ws + (size_t)(8)  * 1024 * 1024);    // 8 MB
  _Float16* Vh = (_Float16*)(ws + (size_t)(16) * 1024 * 1024);    // 8 MB
  float*    AO = (float*)   (ws + (size_t)(24) * 1024 * 1024);    // 16 MB

  dim3 gp(D_MODEL / 64, NQ / 64), bp(128);
  proj_gemm<true><<<gp, bp, 0, stream>>>(Q, Wq, bq, (void*)Qh);
  proj_gemm<true><<<gp, bp, 0, stream>>>(K, Wk, bk, (void*)Kh);
  proj_gemm<true><<<gp, bp, 0, stream>>>(V, Wv, bv, (void*)Vh);

  dim3 ga(NQ / 64, NHEAD);
  attn_kernel<<<ga, bp, 0, stream>>>(Qh, Kh, Vh, bias, AO);

  proj_gemm<false><<<gp, bp, 0, stream>>>(AO, Wo, bo, d_out);
}
